// LSTMTagger_28887950033102
// MI455X (gfx1250) — matrix operand order repaired and checked by reading, not compiled, not run on hardware
//
#include <hip/hip_runtime.h>

// Problem constants (match reference)
#define H    1024
#define B    128
#define T    128
#define COUT 3
#define BH   (B * H)          // 131072
#define TBH  (T * B * H)      // 16777216
#define G3H  (3 * H)          // 3072
#define NBLK 64               // persistent GRU blocks (one per WGP, LDS-resident)
#define WLDS_BYTES (6 * 16 * H * 2)   // 192KB weight slice per block

typedef unsigned short u16b;
typedef unsigned int   u32b;

typedef __attribute__((ext_vector_type(16))) __bf16 v16bf;
typedef __attribute__((ext_vector_type(8)))  float  v8f;

// ---------------- bf16 helpers ----------------
__device__ __forceinline__ float bf2f(u16b h) {
    union { u32b u; float f; } cv; cv.u = ((u32b)h) << 16; return cv.f;
}
__device__ __forceinline__ u16b f2bf(float x) {
    union { float f; u32b u; } cv; cv.f = x;
    u32b u = cv.u;
    u32b r = u + 0x7FFFu + ((u >> 16) & 1u);   // round-to-nearest-even
    return (u16b)(r >> 16);
}

// ---------------- CDNA5 async global->LDS (ASYNCcnt path) ----------------
__device__ __forceinline__ void async_load_b128(unsigned lds_off, const void* gaddr) {
    asm volatile("global_load_async_to_lds_b128 %0, %1, off"
                 :: "v"(lds_off), "v"(gaddr) : "memory");
}
__device__ __forceinline__ void wait_asynccnt0() {
    asm volatile("s_wait_asynccnt 0" ::: "memory");
}

// ---------------- WMMA fragment loaders (base-pointer form, 2x b128 each) ----
// A fragment 16x32 bf16: per-lane base pA = A + (lane&15)*lda + (lane>>4)*8.
// Chunks: [k0 .. k0+7] and [k0+16 .. k0+23] (8 bf16 = 16B each).
__device__ __forceinline__ v16bf load_a16(const u16b* pA, int k0) {
    union { v16bf v; uint4 q[2]; } f;
    f.q[0] = *(const uint4*)(pA + k0);
    f.q[1] = *(const uint4*)(pA + k0 + 16);
    return f.v;
}
// B fragment 32x16 bf16 from [N,K] copy: per-lane base
// pB = Bt + (lane&15)*ldb + (lane>>4)*8; chunks [k0..k0+7] and [k0+16..k0+23].
__device__ __forceinline__ v16bf load_b16(const u16b* pB, int k0) {
    union { v16bf v; uint4 q[2]; } f;
    f.q[0] = *(const uint4*)(pB + k0);
    f.q[1] = *(const uint4*)(pB + k0 + 16);
    return f.v;
}

__device__ __forceinline__ v8f wmma_bf16(v16bf a, v16bf b, v8f c) {
    return __builtin_amdgcn_wmma_f32_16x16x32_bf16(false, a, false, b,
                                                   (short)0, c, false, false);
}

// ---------------- persistent fused GRU (all T steps in one launch) ----------------
// 64 blocks; block owns 16 hidden columns (n0), 192KB weight slice LDS-resident.
// Wave w handles batch rows m0 = 16*w.
// LDS layout (column-interleaved so a lane's 6 gate rows sit within 12KB):
//   row for (tile column c, gate g) at element offset (c*6 + g) * H,
//   g = 0..2: Wih{r,z,n},  g = 3..5: Whh{r,z,n}.
__global__ void __launch_bounds__(256, 1) gru_persistent_kernel(
    const u16b* __restrict__ Xseq,    // [B,T,H] bf16 inputs
    const u16b* __restrict__ H0,      // [B,H] initial state
    u16b* __restrict__ Hist, int hist,// hist=1: write state t -> Hist + t*BH
    u16b* __restrict__ P0, u16b* __restrict__ P1, // ping-pong when hist=0
    const u16b* __restrict__ Wih,     // [3H,H] bf16
    const u16b* __restrict__ Whh,     // [3H,H] bf16
    const float* __restrict__ bih, const float* __restrict__ bhh,
    unsigned* __restrict__ sync_cnt)  // [T] zero-initialized arrival counters
{
    extern __shared__ u16b wlds[];    // 6*16*H bf16 = 192KB
    const int n0   = blockIdx.x << 4;
    const int lane = threadIdx.x & 31;
    const int wave = threadIdx.x >> 5;
    const int m0   = wave << 4;
    const int lr   = lane & 15;
    const int hf   = lane >> 4;

    // ---- one-time async weight preload: global -> LDS, ASYNCcnt tracked ----
    for (int idx = threadIdx.x; idx < 96 * (H / 8); idx += 256) {
        int rowi  = idx / (H / 8);           // 0..95  (= g*16 + r)
        int chunk = idx % (H / 8);           // 16B chunks along K
        int g = rowi >> 4, r = rowi & 15;
        const u16b* src = (g < 3 ? Wih + (size_t)(g * H + n0 + r) * H
                                 : Whh + (size_t)((g - 3) * H + n0 + r) * H)
                          + chunk * 8;
        unsigned dst = (unsigned)(size_t)(&wlds[(size_t)(r * 6 + g) * H + chunk * 8]);
        async_load_b128(dst, src);
    }
    wait_asynccnt0();
    __syncthreads();

    // single per-lane LDS base; six gate rows at +g*H (fits DS offset immediate)
    const u16b* pSb = wlds + (size_t)(lr * 6) * H + hf * 8;

    // per-lane A-row base for the input sequence (invariant except + t*H)
    const u16b* pX = Xseq + (size_t)(m0 + lr) * (T * H) + hf * 8;

    const int col   = n0 + lr;
    const int rbase = m0 + (hf << 3);
    const float br  = bih[col]         + bhh[col];
    const float bz  = bih[H + col]     + bhh[H + col];
    const float bxn = bih[2 * H + col];
    const float bhn = bhh[2 * H + col];

    for (int t = 0; t < T; ++t) {
        const u16b* hin = (t == 0) ? H0
                        : (hist ? Hist + (size_t)(t - 1) * BH
                                : ((t & 1) ? P0 : P1));
        u16b* hout = hist ? Hist + (size_t)t * BH : ((t & 1) ? P1 : P0);

        const u16b* pAx = pX + (size_t)t * H;
        const u16b* pAh = hin + (size_t)(m0 + lr) * H + hf * 8;

        // ---- software-pipelined K loop: prologue loads k=0 fragments ----
        v16bf ax = load_a16(pAx, 0);
        v16bf ah = load_a16(pAh, 0);
        v16bf w0 = load_b16(pSb + 0 * H, 0);
        v16bf w1 = load_b16(pSb + 1 * H, 0);
        v16bf w2 = load_b16(pSb + 2 * H, 0);
        v16bf w3 = load_b16(pSb + 3 * H, 0);
        v16bf w4 = load_b16(pSb + 4 * H, 0);
        v16bf w5 = load_b16(pSb + 5 * H, 0);

        v8f aR = {}, aZ = {}, aXN = {}, aHN = {};
        for (int k0 = 0; k0 < H; k0 += 32) {
            const int kn = (k0 + 32) & (H - 1);   // last iter wraps (harmless)
            v16bf axn = load_a16(pAx, kn);
            v16bf ahn = load_a16(pAh, kn);
            v16bf y0 = load_b16(pSb + 0 * H, kn);
            v16bf y1 = load_b16(pSb + 1 * H, kn);
            v16bf y2 = load_b16(pSb + 2 * H, kn);
            v16bf y3 = load_b16(pSb + 3 * H, kn);
            v16bf y4 = load_b16(pSb + 4 * H, kn);
            v16bf y5 = load_b16(pSb + 5 * H, kn);

            aR  = wmma_bf16(ax, w0, aR);
            aZ  = wmma_bf16(ax, w1, aZ);
            aXN = wmma_bf16(ax, w2, aXN);
            aR  = wmma_bf16(ah, w3, aR);
            aZ  = wmma_bf16(ah, w4, aZ);
            aHN = wmma_bf16(ah, w5, aHN);

            ax = axn; ah = ahn;
            w0 = y0; w1 = y1; w2 = y2; w3 = y3; w4 = y4; w5 = y5;
        }

        union { v8f v; float f[8]; } R, Z, XN, HN;
        R.v = aR; Z.v = aZ; XN.v = aXN; HN.v = aHN;
#pragma unroll
        for (int i = 0; i < 8; ++i) {
            int row = rbase + i;
            float hold = bf2f(hin[(size_t)row * H + col]);
            float rg = 1.f / (1.f + __expf(-(R.f[i] + br)));
            float zg = 1.f / (1.f + __expf(-(Z.f[i] + bz)));
            float ng = tanhf(XN.f[i] + bxn + rg * (HN.f[i] + bhn));
            hout[(size_t)row * H + col] = f2bf((1.f - zg) * ng + zg * hold);
        }

        // ---- device-scope grid barrier (release stores, acquire before next read)
        __syncthreads();
        if (threadIdx.x == 0) {
            __hip_atomic_fetch_add(&sync_cnt[t], 1u, __ATOMIC_RELEASE,
                                   __HIP_MEMORY_SCOPE_AGENT);
            while (__hip_atomic_load(&sync_cnt[t], __ATOMIC_ACQUIRE,
                                     __HIP_MEMORY_SCOPE_AGENT) < (unsigned)NBLK) {
                __builtin_amdgcn_s_sleep(2);
            }
        }
        __syncthreads();
    }
}

// ---------------- generic bf16 NT-GEMM: C = A[MxK] * Bt[NxK]^T ----------------
__global__ void __launch_bounds__(256) gemm_bf16_nt(
    const u16b* __restrict__ A, int lda,
    const u16b* __restrict__ Bt, int ldb,
    int M, int N, int K,
    float* __restrict__ Cf, u16b* __restrict__ Cb,
    const float* __restrict__ bias,
    const float* __restrict__ elemAdd,
    const float* __restrict__ rowAdd, int rowMod,
    int do_tanh)
{
    int lane = threadIdx.x & 31;
    int wave = threadIdx.x >> 5;
    int tile = blockIdx.x * 8 + wave;
    int tilesN = N >> 4;
    int m0 = (tile / tilesN) << 4;
    int n0 = (tile % tilesN) << 4;
    if (m0 >= M) return;
    int lr = lane & 15, hf = lane >> 4;

    const u16b* pA = A  + (size_t)(m0 + lr) * lda + hf * 8;
    const u16b* pB = Bt + (size_t)(n0 + lr) * ldb + hf * 8;

    v8f acc = {};
    for (int k0 = 0; k0 < K; k0 += 32) {
        v16bf af = load_a16(pA, k0);
        v16bf bf = load_b16(pB, k0);
        acc = wmma_bf16(af, bf, acc);
    }

    union { v8f v; float f[8]; } r; r.v = acc;
    int col   = n0 + lr;
    int rbase = m0 + (hf << 3);
    float bcol = bias ? bias[col] : 0.f;
#pragma unroll
    for (int i = 0; i < 8; ++i) {
        int row = rbase + i;
        float v = r.f[i] + bcol;
        if (elemAdd) v += elemAdd[(size_t)row * N + col];
        if (rowAdd)  v += rowAdd[(size_t)(row % rowMod) * N + col];
        if (do_tanh) v = tanhf(v);
        if (Cf) Cf[(size_t)row * N + col] = v;
        if (Cb) Cb[(size_t)row * N + col] = f2bf(v);
    }
}

// ---------------- attention small kernels ----------------
__global__ void __launch_bounds__(256) attn_scores_kernel(
    const u16b* __restrict__ Mb,      // [T*B, H], row r = t*B + b
    const float* __restrict__ walpha, // [H]
    float* __restrict__ scores)       // [B, T]
{
    int lane = threadIdx.x & 31;
    int wave = threadIdx.x >> 5;
    int r = blockIdx.x * 8 + wave;
    const u16b* row = Mb + (size_t)r * H;
    float acc = 0.f;
    for (int h = lane; h < H; h += 32) acc += bf2f(row[h]) * walpha[h];
#pragma unroll
    for (int off = 16; off; off >>= 1) acc += __shfl_xor(acc, off, 32);
    if (lane == 0) {
        int t = r >> 7, b = r & 127;
        scores[b * T + t] = acc;
    }
}

__global__ void attn_softmax_kernel(const float* __restrict__ scores,
                                    float* __restrict__ alpha)
{
    __shared__ float sh[T];
    int b = blockIdx.x, t = threadIdx.x;
    float s = scores[b * T + t];
    sh[t] = s; __syncthreads();
    for (int off = 64; off; off >>= 1) {
        if (t < off) sh[t] = fmaxf(sh[t], sh[t + off]);
        __syncthreads();
    }
    float m = sh[0]; __syncthreads();
    float e = __expf(s - m);
    sh[t] = e; __syncthreads();
    for (int off = 64; off; off >>= 1) {
        if (t < off) sh[t] += sh[t + off];
        __syncthreads();
    }
    alpha[b * T + t] = e / sh[0];
}

__global__ void __launch_bounds__(256) attn_r_kernel(
    const float* __restrict__ alpha,  // [B,T]
    const u16b* __restrict__ Y,       // o_p [T,B,H] bf16
    u16b* __restrict__ rb)            // [B,H] bf16
{
    int idx = blockIdx.x * 256 + threadIdx.x;
    int b = idx >> 10, h = idx & 1023;
    float acc = 0.f;
    for (int t = 0; t < T; ++t)
        acc += alpha[b * T + t] * bf2f(Y[(size_t)t * BH + (size_t)b * H + h]);
    rb[idx] = f2bf(acc);
}

__global__ void final_logits_kernel(const float* __restrict__ hstar,
                                    const float* __restrict__ out_w,
                                    const float* __restrict__ out_b,
                                    float* __restrict__ out)
{
    int b = threadIdx.x;   // 128 threads
    float l[COUT];
#pragma unroll
    for (int c = 0; c < COUT; ++c) {
        float acc = out_b[c];
        const float* w  = out_w + (size_t)c * H;
        const float* hs = hstar + (size_t)b * H;
        for (int h = 0; h < H; ++h) acc += hs[h] * w[h];
        l[c] = tanhf(acc);
    }
    float m = fmaxf(l[0], fmaxf(l[1], l[2]));
    float s = __expf(l[0] - m) + __expf(l[1] - m) + __expf(l[2] - m);
    float lse = m + __logf(s);
#pragma unroll
    for (int c = 0; c < COUT; ++c) out[b * COUT + c] = l[c] - lse;
}

// ---------------- conversion / init kernels ----------------
__global__ void cvt_f32_bf16_kernel(const float* __restrict__ in,
                                    u16b* __restrict__ out, int n) {
    int i = blockIdx.x * 256 + threadIdx.x;
    if (i < n) out[i] = f2bf(in[i]);
}
__global__ void cvt_tr_bf16_kernel(const float* __restrict__ in,
                                   u16b* __restrict__ out, int R, int Cc) {
    int i = blockIdx.x * 256 + threadIdx.x;
    if (i < R * Cc) {
        int r = i / Cc, c = i % Cc;
        out[(size_t)c * R + r] = f2bf(in[i]);
    }
}
__global__ void zero_bf16_kernel(u16b* __restrict__ p, int n) {
    int i = blockIdx.x * 256 + threadIdx.x;
    if (i < n) p[i] = 0;
}
__global__ void zero_u32_kernel(unsigned* __restrict__ p, int n) {
    int i = blockIdx.x * 256 + threadIdx.x;
    if (i < n) p[i] = 0u;
}

// ---------------- host orchestration ----------------
extern "C" void kernel_launch(void* const* d_in, const int* in_sizes, int n_in,
                              void* d_out, int out_size, void* d_ws, size_t ws_size,
                              hipStream_t stream)
{
    (void)in_sizes; (void)n_in; (void)out_size; (void)ws_size;

    const float* premise    = (const float*)d_in[0];
    const float* hypothesis = (const float*)d_in[1];
    const float* p_Wih = (const float*)d_in[2];
    const float* p_Whh = (const float*)d_in[3];
    const float* p_bih = (const float*)d_in[4];
    const float* p_bhh = (const float*)d_in[5];
    const float* h_Wih = (const float*)d_in[6];
    const float* h_Whh = (const float*)d_in[7];
    const float* h_bih = (const float*)d_in[8];
    const float* h_bhh = (const float*)d_in[9];
    const float* W_y     = (const float*)d_in[10];
    const float* W_h     = (const float*)d_in[11];
    const float* W_alpha = (const float*)d_in[12];
    const float* W_x     = (const float*)d_in[13];
    const float* W_p     = (const float*)d_in[14];
    const float* out_w   = (const float*)d_in[15];
    const float* out_b   = (const float*)d_in[16];
    float* out = (float*)d_out;

    // workspace carve-up (deterministic)
    char* wsp = (char*)d_ws;
    size_t off = 0;
    auto alloc = [&](size_t bytes) -> char* {
        char* p = wsp + off;
        off = (off + bytes + 255) & ~(size_t)255;
        return p;
    };
    u16b* prem_b = (u16b*)alloc((size_t)TBH * 2);
    u16b* hyp_b  = (u16b*)alloc((size_t)TBH * 2);
    u16b* Wih_p  = (u16b*)alloc((size_t)G3H * H * 2);
    u16b* Whh_p  = (u16b*)alloc((size_t)G3H * H * 2);
    u16b* Wih_h  = (u16b*)alloc((size_t)G3H * H * 2);
    u16b* Whh_h  = (u16b*)alloc((size_t)G3H * H * 2);
    u16b* Wty    = (u16b*)alloc((size_t)H * H * 2);
    u16b* Wth    = (u16b*)alloc((size_t)H * H * 2);
    u16b* Wtp    = (u16b*)alloc((size_t)H * H * 2);
    u16b* Wtx    = (u16b*)alloc((size_t)H * H * 2);
    u16b* o_p    = (u16b*)alloc((size_t)TBH * 2);   // premise states [T,B,H]
    u16b* zbuf   = (u16b*)alloc((size_t)BH * 2);
    u16b* hBuf0  = (u16b*)alloc((size_t)BH * 2);
    u16b* hBuf1  = (u16b*)alloc((size_t)BH * 2);
    float* WhB   = (float*)alloc((size_t)BH * 4);
    float* scores = (float*)alloc((size_t)B * T * 4);
    float* alphaB = (float*)alloc((size_t)B * T * 4);
    u16b* r_b    = (u16b*)alloc((size_t)BH * 2);
    float* tmpC  = (float*)alloc((size_t)BH * 4);
    float* hstar = (float*)alloc((size_t)BH * 4);
    unsigned* syncc = (unsigned*)alloc((size_t)2 * T * 4);
    u16b* Mbuf   = hyp_b;   // hypothesis bf16 input dead after hyp GRU; reuse

    // 1) conversions + barrier-counter init
    cvt_f32_bf16_kernel<<<(TBH + 255) / 256, 256, 0, stream>>>(premise, prem_b, TBH);
    cvt_f32_bf16_kernel<<<(TBH + 255) / 256, 256, 0, stream>>>(hypothesis, hyp_b, TBH);
    cvt_f32_bf16_kernel<<<(G3H * H + 255) / 256, 256, 0, stream>>>(p_Wih, Wih_p, G3H * H);
    cvt_f32_bf16_kernel<<<(G3H * H + 255) / 256, 256, 0, stream>>>(p_Whh, Whh_p, G3H * H);
    cvt_f32_bf16_kernel<<<(G3H * H + 255) / 256, 256, 0, stream>>>(h_Wih, Wih_h, G3H * H);
    cvt_f32_bf16_kernel<<<(G3H * H + 255) / 256, 256, 0, stream>>>(h_Whh, Whh_h, G3H * H);
    cvt_tr_bf16_kernel<<<(H * H + 255) / 256, 256, 0, stream>>>(W_y, Wty, H, H);
    cvt_tr_bf16_kernel<<<(H * H + 255) / 256, 256, 0, stream>>>(W_h, Wth, H, H);
    cvt_tr_bf16_kernel<<<(H * H + 255) / 256, 256, 0, stream>>>(W_p, Wtp, H, H);
    cvt_tr_bf16_kernel<<<(H * H + 255) / 256, 256, 0, stream>>>(W_x, Wtx, H, H);
    zero_bf16_kernel<<<(BH + 255) / 256, 256, 0, stream>>>(zbuf, BH);
    zero_u32_kernel<<<1, 256, 0, stream>>>(syncc, 2 * T);

    // 2) premise GRU: single persistent launch, weights LDS-resident,
    //    per-step states stream into o_p[t]
    gru_persistent_kernel<<<NBLK, 256, WLDS_BYTES, stream>>>(
        prem_b, zbuf, o_p, 1, nullptr, nullptr,
        Wih_p, Whh_p, p_bih, p_bhh, syncc);

    // 3) hypothesis GRU: seeded with premise final state, ping-pong buffers
    gru_persistent_kernel<<<NBLK, 256, WLDS_BYTES, stream>>>(
        hyp_b, o_p + (size_t)(T - 1) * BH, nullptr, 0, hBuf0, hBuf1,
        Wih_h, Whh_h, h_bih, h_bhh, syncc + T);
    u16b* h_last = hBuf1;   // t = 127 (odd) wrote P1

    // 4) attention
    gemm_bf16_nt<<<(8 * 64) / 8, 256, 0, stream>>>(h_last, H, Wth, H, B, H, H,
                                                   WhB, nullptr, nullptr, nullptr,
                                                   nullptr, 1, 0);
    gemm_bf16_nt<<<((T * B / 16) * (H / 16)) / 8, 256, 0, stream>>>(
        o_p, H, Wty, H, T * B, H, H,
        nullptr, Mbuf, nullptr, nullptr, WhB, B, 1);
    attn_scores_kernel<<<(T * B) / 8, 256, 0, stream>>>(Mbuf, W_alpha, scores);
    attn_softmax_kernel<<<B, T, 0, stream>>>(scores, alphaB);
    attn_r_kernel<<<BH / 256, 256, 0, stream>>>(alphaB, o_p, r_b);

    // 5) h* = tanh(r @ W_p + h_last @ W_x)
    gemm_bf16_nt<<<(8 * 64) / 8, 256, 0, stream>>>(r_b, H, Wtp, H, B, H, H,
                                                   tmpC, nullptr, nullptr, nullptr,
                                                   nullptr, 1, 0);
    gemm_bf16_nt<<<(8 * 64) / 8, 256, 0, stream>>>(h_last, H, Wtx, H, B, H, H,
                                                   hstar, nullptr, nullptr, tmpC,
                                                   nullptr, 1, 1);

    // 6) logits + log_softmax
    final_logits_kernel<<<1, B, 0, stream>>>(hstar, out_w, out_b, out);
}
